// AttentionHead_67619965108547
// MI455X (gfx1250) — matrix operand order repaired and checked by reading, not compiled, not run on hardware
//
#include <hip/hip_runtime.h>
#include <hip/hip_bf16.h>

typedef __bf16 bf16_t;
typedef __attribute__((ext_vector_type(16))) __bf16 v16bf;
typedef __attribute__((ext_vector_type(8)))  __bf16 v8bf;
typedef __attribute__((ext_vector_type(8)))  float  v8f;

#define DM   1024
#define DH   128
#define SEQ  4096
#define NB   4
// scale folded into exp2 domain: (1/sqrt(128)) * log2(e)
#define SCALE_QK2 (0.088388347648318447f * 1.4426950408889634f)
#define NEG_BIG (-3.0e38f)

// ---------------------------------------------------------------------------
// WMMA fragment loaders (CDNA5 wave32 layouts, 16x16x32 bf16)
//   A (16x32, MxK): lane -> row = lane&15; k = {h*8..+7, h*8+16..+23}, h=lane>>4
//   B (32x16, KxN): lane -> col = lane&15; k = {h*8..+7, h*8+16..+23}, h=lane>>4
// ---------------------------------------------------------------------------
__device__ __forceinline__ v16bf ld_frag_a(const bf16_t* base, int stride) {
  const int lane = threadIdx.x & 31;
  const bf16_t* p = base + (lane & 15) * stride + (lane >> 4) * 8;
  v8bf lo = *(const v8bf*)(p);
  v8bf hi = *(const v8bf*)(p + 16);
  v16bf r;
#pragma unroll
  for (int i = 0; i < 8; ++i) { r[i] = lo[i]; r[i + 8] = hi[i]; }
  return r;
}

__device__ __forceinline__ v16bf ld_frag_b(const bf16_t* base, int stride) {
  const int lane = threadIdx.x & 31;
  const bf16_t* p = base + (lane & 15) * stride + (lane >> 4) * 8;
  v8bf lo = *(const v8bf*)(p);
  v8bf hi = *(const v8bf*)(p + 16);
  v16bf r;
#pragma unroll
  for (int i = 0; i < 8; ++i) { r[i] = lo[i]; r[i + 8] = hi[i]; }
  return r;
}

__device__ __forceinline__ v8f wmma_bf16(v16bf a, v16bf b, v8f c) {
  return __builtin_amdgcn_wmma_f32_16x16x32_bf16(false, a, false, b, (short)0, c,
                                                 false, false);
}

// ---------------------------------------------------------------------------
// CDNA5 async global->LDS copies (ASYNCcnt path), used for K/V double buffer.
// vdst VGPR holds the per-lane LDS byte offset (relative to wave LDS base).
// ---------------------------------------------------------------------------
__device__ __forceinline__ unsigned lds_off_of(const void* p) {
  return (unsigned)(uintptr_t)(__attribute__((address_space(3))) const void*)p;
}

__device__ __forceinline__ void async_b128(unsigned ldsoff, const bf16_t* g) {
  asm volatile("global_load_async_to_lds_b128 %0, %1, off"
               :: "v"(ldsoff), "v"((unsigned long long)(uintptr_t)g)
               : "memory");
}

__device__ __forceinline__ void wait_async_le16() {
  asm volatile("s_wait_asynccnt 0x10" ::: "memory");
}
__device__ __forceinline__ void wait_async_0() {
  asm volatile("s_wait_asynccnt 0x0" ::: "memory");
}

// ---------------------------------------------------------------------------
// Projection GEMM: Y = X[M x 1024] * W^T[1024 x 128], output bf16.
// transpose_out=0 : Y[i][n]      (natural, for Q and K)
// transpose_out=1 : Y[b][n][s]   (V transposed per batch, for the PV WMMA)
// ---------------------------------------------------------------------------
#define XS 72
#define WSS 72

__global__ void __launch_bounds__(256)
proj_kernel(const float* __restrict__ X, const float* __restrict__ W,
            bf16_t* __restrict__ Y, int transpose_out)
{
  __shared__ bf16_t Xs[64 * XS];
  __shared__ bf16_t Wsh[128 * WSS];

  const int t    = threadIdx.x;
  const int wave = t >> 5;
  const int lane = t & 31;
  const int ln   = lane & 15;
  const int hi8  = (lane >> 4) * 8;
  const int row0 = blockIdx.x * 64;

  const int rtile = wave & 3;
  const int nbase = (wave >> 2) * 4;

  v8f acc[4];
#pragma unroll
  for (int i = 0; i < 4; ++i)
#pragma unroll
    for (int j = 0; j < 8; ++j) acc[i][j] = 0.0f;

  for (int kk = 0; kk < DM; kk += 64) {
    { // stage X (64 x 64) fp32 -> bf16
      int r = t >> 2, c0 = (t & 3) * 16;
      const float4* src = (const float4*)(X + (size_t)(row0 + r) * DM + kk + c0);
      bf16_t* dst = &Xs[r * XS + c0];
#pragma unroll
      for (int i = 0; i < 4; ++i) {
        float4 v = src[i];
        dst[i * 4 + 0] = (bf16_t)v.x; dst[i * 4 + 1] = (bf16_t)v.y;
        dst[i * 4 + 2] = (bf16_t)v.z; dst[i * 4 + 3] = (bf16_t)v.w;
      }
    }
    { // stage W (128 x 64) fp32 -> bf16
      int r = t >> 1, c0 = (t & 1) * 32;
      const float4* src = (const float4*)(W + (size_t)r * DM + kk + c0);
      bf16_t* dst = &Wsh[r * WSS + c0];
#pragma unroll
      for (int i = 0; i < 8; ++i) {
        float4 v = src[i];
        dst[i * 4 + 0] = (bf16_t)v.x; dst[i * 4 + 1] = (bf16_t)v.y;
        dst[i * 4 + 2] = (bf16_t)v.z; dst[i * 4 + 3] = (bf16_t)v.w;
      }
    }
    __syncthreads();

#pragma unroll
    for (int kc = 0; kc < 2; ++kc) {
      v16bf a = ld_frag_a(&Xs[rtile * 16 * XS + kc * 32], XS);
#pragma unroll
      for (int i = 0; i < 4; ++i) {
        v16bf b = ld_frag_b(&Wsh[(nbase + i) * 16 * WSS + kc * 32], WSS);
        acc[i] = wmma_bf16(a, b, acc[i]);
      }
    }
    __syncthreads();
  }

  if (!transpose_out) {
#pragma unroll
    for (int i = 0; i < 4; ++i) {
      int col = (nbase + i) * 16 + ln;
#pragma unroll
      for (int r = 0; r < 8; ++r) {
        int irow = row0 + rtile * 16 + r + hi8;
        Y[(size_t)irow * DH + col] = (bf16_t)acc[i][r];
      }
    }
  } else {
    int bb = row0 / SEQ;
    int s0 = (row0 % SEQ) + rtile * 16 + hi8;
#pragma unroll
    for (int i = 0; i < 4; ++i) {
      int col = (nbase + i) * 16 + ln;
      v8bf pk;
#pragma unroll
      for (int r = 0; r < 8; ++r) pk[r] = (bf16_t)acc[i][r];
      *(v8bf*)&Y[((size_t)bb * DH + col) * SEQ + s0] = pk;
    }
  }
}

// ---------------------------------------------------------------------------
// Causal flash attention, double-buffered async K/V staging.
// 128 threads = 4 waves; wave w owns query rows [i0+16w, i0+16w+16).
// ---------------------------------------------------------------------------
#define KSS 136   // 64x128 K tile, padded row stride (bf16)
#define VSS 72    // 128x64 V^T tile
#define PSS 72    // per-wave 16x64 P patch

// issue the 16 per-thread async b128 copies for one K/V tile pair
__device__ __forceinline__ void stage_tiles(const bf16_t* Kg, const bf16_t* Vg,
                                            unsigned ksOff, unsigned vsOff,
                                            int j0, int t)
{
  { // K tile (64 x 128): 2 threads per row
    int r = t >> 1, c0 = (t & 1) * 64;
    const bf16_t* src = Kg + (size_t)(j0 + r) * DH + c0;
    unsigned dst = ksOff + (unsigned)(r * KSS + c0) * 2u;
#pragma unroll
    for (int i = 0; i < 8; ++i) async_b128(dst + 16u * i, src + 8 * i);
  }
  { // V^T tile (128 x 64): 1 thread per row
    const bf16_t* src = Vg + (size_t)t * SEQ + j0;
    unsigned dst = vsOff + (unsigned)(t * VSS) * 2u;
#pragma unroll
    for (int i = 0; i < 8; ++i) async_b128(dst + 16u * i, src + 8 * i);
  }
}

__global__ void __launch_bounds__(128)
attn_kernel(const bf16_t* __restrict__ Q, const bf16_t* __restrict__ K,
            const bf16_t* __restrict__ Vt, float* __restrict__ O)
{
  __shared__ bf16_t Ks[2][64 * KSS];
  __shared__ bf16_t Vs[2][128 * VSS];
  __shared__ bf16_t Ps[4 * 16 * PSS];

  const int t    = threadIdx.x;
  const int wave = t >> 5;
  const int lane = t & 31;
  const int ln   = lane & 15;
  const int hi8  = (lane >> 4) * 8;

  const int nblk  = SEQ / 64;
  const int b     = blockIdx.x / nblk;
  // reversed order: longest (diagonal-heavy) blocks launch first
  const int i0    = (nblk - 1 - (blockIdx.x % nblk)) * 64;
  const int ibase = i0 + wave * 16;

  const bf16_t* Qg = Q  + ((size_t)b * SEQ + ibase) * DH;
  const bf16_t* Kg = K  + (size_t)b * SEQ * DH;
  const bf16_t* Vg = Vt + (size_t)b * DH * SEQ;

  unsigned ksOff[2] = { lds_off_of(&Ks[0][0]), lds_off_of(&Ks[1][0]) };
  unsigned vsOff[2] = { lds_off_of(&Vs[0][0]), lds_off_of(&Vs[1][0]) };

  v16bf Qa[4];
#pragma unroll
  for (int c = 0; c < 4; ++c) Qa[c] = ld_frag_a(Qg + c * 32, DH);

  v8f Oc[8];
#pragma unroll
  for (int i = 0; i < 8; ++i)
#pragma unroll
    for (int j = 0; j < 8; ++j) Oc[i][j] = 0.0f;

  float m[8], l[8];
#pragma unroll
  for (int r = 0; r < 8; ++r) { m[r] = NEG_BIG; l[r] = 0.0f; }

  bf16_t* pw = &Ps[wave * 16 * PSS];

  const int ntiles = i0 / 64 + 1;     // causal: up to the diagonal block
  stage_tiles(Kg, Vg, ksOff[0], vsOff[0], 0, t);   // prologue

  for (int tile = 0; tile < ntiles; ++tile) {
    const int cur  = tile & 1;
    const int j0   = tile * 64;
    const bool more = (tile + 1) < ntiles;
    if (more) stage_tiles(Kg, Vg, ksOff[cur ^ 1], vsOff[cur ^ 1], j0 + 64, t);
    if (more) wait_async_le16(); else wait_async_0();
    __syncthreads();

    const bf16_t* KsC = &Ks[cur][0];
    const bf16_t* VsC = &Vs[cur][0];

    // S = Q K^T (16 x 64 per wave), already in exp2 domain after scaling
    v8f Sv[4];
#pragma unroll
    for (int nt = 0; nt < 4; ++nt) {
#pragma unroll
      for (int j = 0; j < 8; ++j) Sv[nt][j] = 0.0f;
#pragma unroll
      for (int kc = 0; kc < 4; ++kc) {
        v16bf bk = ld_frag_b(KsC + nt * 16 * KSS + kc * 32, KSS);
        Sv[nt] = wmma_bf16(Qa[kc], bk, Sv[nt]);
      }
    }

    // scale (exp2 domain) + causal mask
    const bool need_mask = (j0 + 63) > ibase;
#pragma unroll
    for (int nt = 0; nt < 4; ++nt)
#pragma unroll
      for (int r = 0; r < 8; ++r) {
        float s = Sv[nt][r] * SCALE_QK2;
        if (need_mask) {
          int col = j0 + nt * 16 + ln;
          int row = ibase + r + hi8;
          if (col > row) s = NEG_BIG;
        }
        Sv[nt][r] = s;
      }

    // online softmax in exp2 domain: row stats via 16-lane xor reductions
    float mnew[8], alpha[8];
#pragma unroll
    for (int r = 0; r < 8; ++r) {
      float v = Sv[0][r];
#pragma unroll
      for (int nt = 1; nt < 4; ++nt) v = fmaxf(v, Sv[nt][r]);
      v = fmaxf(v, __shfl_xor(v, 1, 32));
      v = fmaxf(v, __shfl_xor(v, 2, 32));
      v = fmaxf(v, __shfl_xor(v, 4, 32));
      v = fmaxf(v, __shfl_xor(v, 8, 32));
      mnew[r]  = fmaxf(m[r], v);
      alpha[r] = exp2f(m[r] - mnew[r]);
      m[r] = mnew[r];
    }
#pragma unroll
    for (int r = 0; r < 8; ++r) {
      float s = 0.0f;
#pragma unroll
      for (int nt = 0; nt < 4; ++nt) {
        float p = exp2f(Sv[nt][r] - mnew[r]);
        Sv[nt][r] = p;
        s += p;
      }
      s += __shfl_xor(s, 1, 32);
      s += __shfl_xor(s, 2, 32);
      s += __shfl_xor(s, 4, 32);
      s += __shfl_xor(s, 8, 32);
      l[r] = l[r] * alpha[r] + s;
    }

    // rescale running O
#pragma unroll
    for (int dt = 0; dt < 8; ++dt)
#pragma unroll
      for (int r = 0; r < 8; ++r) Oc[dt][r] *= alpha[r];

    // P (D-layout) -> wave-private LDS (row-major A-layout source)
#pragma unroll
    for (int nt = 0; nt < 4; ++nt)
#pragma unroll
      for (int r = 0; r < 8; ++r)
        pw[(r + hi8) * PSS + nt * 16 + ln] = (bf16_t)Sv[nt][r];

    // O += P V  (A = P 16x64, B = V^T tiles)
    v16bf Pa0 = ld_frag_a(pw, PSS);
    v16bf Pa1 = ld_frag_a(pw + 32, PSS);
#pragma unroll
    for (int dt = 0; dt < 8; ++dt) {
      v16bf b0 = ld_frag_b(VsC + dt * 16 * VSS, VSS);
      Oc[dt] = wmma_bf16(Pa0, b0, Oc[dt]);
      v16bf b1 = ld_frag_b(VsC + dt * 16 * VSS + 32, VSS);
      Oc[dt] = wmma_bf16(Pa1, b1, Oc[dt]);
    }
    __syncthreads();   // protect buffers before next async overwrite
  }

  float* outp = O + ((size_t)b * SEQ + ibase) * DH;
#pragma unroll
  for (int r = 0; r < 8; ++r) {
    float inv = 1.0f / l[r];
#pragma unroll
    for (int dt = 0; dt < 8; ++dt)
      outp[(size_t)(r + hi8) * DH + dt * 16 + ln] = Oc[dt][r] * inv;
  }
}

// ---------------------------------------------------------------------------
extern "C" void kernel_launch(void* const* d_in, const int* in_sizes, int n_in,
                              void* d_out, int out_size, void* d_ws, size_t ws_size,
                              hipStream_t stream) {
  const float* q  = (const float*)d_in[0];
  const float* k  = (const float*)d_in[1];
  const float* v  = (const float*)d_in[2];
  const float* Wq = (const float*)d_in[3];
  const float* Wk = (const float*)d_in[4];
  const float* Wv = (const float*)d_in[5];
  float* out = (float*)d_out;

  const size_t M = (size_t)NB * SEQ;          // 16384 rows
  bf16_t* Qb = (bf16_t*)d_ws;                 // [M][128]       bf16
  bf16_t* Kb = Qb + M * DH;                   // [M][128]       bf16
  bf16_t* Vb = Kb + M * DH;                   // [B][128][S]    bf16 (transposed)

  proj_kernel<<<(int)(M / 64), 256, 0, stream>>>(q, Wq, Qb, 0);
  proj_kernel<<<(int)(M / 64), 256, 0, stream>>>(k, Wk, Kb, 0);
  proj_kernel<<<(int)(M / 64), 256, 0, stream>>>(v, Wv, Vb, 1);
  attn_kernel<<<NB * (SEQ / 64), 128, 0, stream>>>(Qb, Kb, Vb, out);
}
